// SmallSMBlock3D_44753559224603
// MI455X (gfx1250) — matrix operand order repaired and checked by reading, not compiled, not run on hardware
//
#include <hip/hip_runtime.h>
#include <stdint.h>

typedef __attribute__((ext_vector_type(16))) __bf16 v16bf;
typedef __attribute__((ext_vector_type(8)))  float  v8f;

static constexpr int NN     = 128;  // cube edge
static constexpr int IPITCH = 132;  // image row pitch (130 used + pad)
static constexpr int XPITCH = 132;  // x row pitch
static constexpr int JT     = 2;    // j-values per block
static constexpr int DQ     = JT + 2;  // staged dq-extent per (c,dp): 4
static constexpr int IROWS  = 9 * DQ;  // 36 image rows
static constexpr int XROWS  = 3 * DQ;  // 12 x rows

// ---------------------------------------------------------------------------
// Prep kernel (1 wave, once per launch): per-lane bf16 A fragments
// (CDNA5 16-bit A layout) for W' = [27x81 weights | bias col at g=81 | 0],
// M padded to 32, K padded to 96. Rolled loops -> tiny static footprint.
//   lane l, fragment (mt,s): 16 bf16 at byte offset l*192 + (mt*3+s)*32
// ---------------------------------------------------------------------------
__global__ void prep_afrag(const float* __restrict__ wts,
                           const float* __restrict__ bias,
                           unsigned short* __restrict__ aws) {
    const int l  = threadIdx.x;          // 0..31
    const int ha = (l < 16) ? 0 : 8;
    #pragma unroll 1
    for (int mt = 0; mt < 2; ++mt) {
        const int m = mt * 16 + (l & 15);
        #pragma unroll 1
        for (int s = 0; s < 3; ++s) {
            #pragma unroll 1
            for (int e = 0; e < 16; ++e) {
                const int g = 32 * s + ha + ((e < 8) ? e : e + 8);
                float v = 0.f;
                if (m < 27) {
                    if (g < 81)       v = wts[m * 81 + g];
                    else if (g == 81) v = bias[m];
                }
                union { __bf16 b; unsigned short u; } cv;
                cv.b = (__bf16)v;
                aws[l * 96 + (mt * 3 + s) * 16 + e] = cv.u;
            }
        }
    }
}

// ---------------------------------------------------------------------------
// B fragment per CDNA5 16-bit B layout: lane<16 -> K 0..7,16..23 (HB=0),
// lane>=16 -> K 8..15,24..31 (HB=16). LDS rows are (c,dp,dq~) with stride-4 dq~,
// so a j-offset is a uniform +IPITCH pointer shift. All offsets immediate.
// g==81 is the bias column (1.0); g>81 is K padding (0).
// ---------------------------------------------------------------------------
template<int HB>
__device__ __forceinline__ void assemble_b(const float* __restrict__ col, v16bf b[3]) {
    #pragma unroll
    for (int s = 0; s < 3; ++s) {
        #pragma unroll
        for (int e = 0; e < 16; ++e) {
            const int g = 32 * s + HB / 2 + ((e < 8) ? e : e + 8);
            if (g < 81) {
                const int c  = g / 27;
                const int dp = (g - c * 27) / 9;
                const int dq = (g % 9) / 3;
                const int dr = g % 3;
                const int rr = (c * 3 + dp) * DQ + dq;
                b[s][e] = (__bf16)col[rr * IPITCH + dr];
            } else if (g == 81) {
                b[s][e] = (__bf16)1.0f;      // bias column
            } else {
                b[s][e] = (__bf16)0.0f;      // K padding
            }
        }
    }
}

// Fused stencil combine (bias already in acc via the g=81 column).
// D layout: lane l, VGPR r -> channel HA + 16*mt + r, voxel l%16.
template<int HA>
__device__ __forceinline__ float combine(const v8f acc[2],
                                         const float* __restrict__ xcol) {
    float t = 0.f;
    #pragma unroll
    for (int mt = 0; mt < 2; ++mt) {
        #pragma unroll
        for (int r = 0; r < 8; ++r) {
            const int kk = HA + 16 * mt + r;
            if (kk < 27) {
                const int dp  = kk / 9;
                const int rem = kk - dp * 9;
                const int dq  = rem / 3, dr = rem - dq * 3;
                t += acc[mt][r] * xcol[(dp * DQ + dq) * XPITCH + dr];
            }
        }
    }
    return t;
}

// ---------------------------------------------------------------------------
// Main kernel: one workgroup = (i, j0..j0+1) x 128 voxels (256 outputs);
// 8 wave32, each owns one 16-voxel column-tile and loops over the 2 j's:
//   K[27x128] = W'[27x96] * im2col'(img)[96x128]   (bias folded in)
// fused with the 27-term shifted-x combine; only y is written.
// ---------------------------------------------------------------------------
__global__ __launch_bounds__(256) void sm_block3d_wmma(
    const float* __restrict__ img,          // (3,128,128,128)
    const float* __restrict__ x,            // (1,1,128,128,128)
    const uint4* __restrict__ aws,          // prebuilt A fragments (bf16)
    float* __restrict__ out)                // (1,1,128,128,128)
{
    __shared__ float lds_img[IROWS * IPITCH];
    __shared__ float lds_x  [XROWS * XPITCH];

    const int j0   = blockIdx.x * JT;  // H base
    const int i    = blockIdx.y;       // D
    const int tid  = threadIdx.x;
    const int lane = tid & 31;
    const int wv   = tid >> 5;

    // ---- stage image rows (rr wave-uniform -> scalar row math) ----
    for (int rr = wv; rr < IROWS; rr += 8) {
        const int c = rr / (3 * DQ), rem = rr - c * 3 * DQ;
        const int dp = rem / DQ, dqq = rem - dp * DQ;
        const int z = i + dp - 1, y = j0 + dqq - 1;
        const bool rowok = (z >= 0) & (z < NN) & (y >= 0) & (y < NN);
        const float* src = img + ((c * NN + z) * NN + y) * NN;
        for (int t = lane; t < IPITCH; t += 32) {
            const int w = t - 1;
            lds_img[rr * IPITCH + t] = (rowok && w >= 0 && w < NN) ? src[w] : 0.f;
        }
    }
    // ---- stage x rows ----
    for (int rr = wv; rr < XROWS; rr += 8) {
        const int dp = rr / DQ, dqq = rr - dp * DQ;
        const int z = i + dp - 1, y = j0 + dqq - 1;
        const bool rowok = (z >= 0) & (z < NN) & (y >= 0) & (y < NN);
        const float* src = x + (z * NN + y) * NN;
        for (int t = lane; t < XPITCH; t += 32) {
            const int w = t - 1;
            lds_x[rr * XPITCH + t] = (rowok && w >= 0 && w < NN) ? src[w] : 0.f;
        }
    }

    // ---- load prebuilt A fragments: 12x global_load_b128 (L2 broadcast) ----
    union AFrag { uint4 q[2]; v16bf v; };
    AFrag af[2][3];
    {
        const uint4* asrc = aws + lane * 12;     // 12 uint4 = 192 B per lane
        #pragma unroll
        for (int mt = 0; mt < 2; ++mt)
            #pragma unroll
            for (int s = 0; s < 3; ++s) {
                af[mt][s].q[0] = asrc[(mt * 3 + s) * 2 + 0];
                af[mt][s].q[1] = asrc[(mt * 3 + s) * 2 + 1];
            }
    }
    __syncthreads();

    const int nloc = lane & 15;
    const int ncol = wv * 16 + nloc;       // voxel l

    #pragma unroll 1
    for (int it = 0; it < JT; ++it) {      // j_off = uniform LDS pointer shift
        v16bf bfrag[3];
        {
            const float* icol = lds_img + it * IPITCH + ncol;
            if (lane < 16) assemble_b<0>(icol, bfrag);
            else           assemble_b<16>(icol, bfrag);
        }

        // ---- implicit-GEMM conv (EXEC all-ones here) ----
        v8f acc[2] = {};
        #pragma unroll
        for (int s = 0; s < 3; ++s) {
            acc[0] = __builtin_amdgcn_wmma_f32_16x16x32_bf16(
                false, af[0][s].v, false, bfrag[s], (short)0, acc[0], false, false);
            acc[1] = __builtin_amdgcn_wmma_f32_16x16x32_bf16(
                false, af[1][s].v, false, bfrag[s], (short)0, acc[1], false, false);
        }

        // ---- fused combine + cross-half reduction ----
        const float* xcol = lds_x + it * XPITCH + ncol;
        float total = (lane < 16) ? combine<0>(acc, xcol)
                                  : combine<8>(acc, xcol);
        total += __shfl_xor(total, 16);
        if (lane < 16)
            out[(i * NN + j0 + it) * NN + ncol] = total;
    }
}

extern "C" void kernel_launch(void* const* d_in, const int* in_sizes, int n_in,
                              void* d_out, int out_size, void* d_ws, size_t ws_size,
                              hipStream_t stream) {
    (void)in_sizes; (void)n_in; (void)out_size; (void)ws_size;
    const float* img  = (const float*)d_in[0];
    const float* x    = (const float*)d_in[1];
    const float* wts  = (const float*)d_in[2];
    const float* bias = (const float*)d_in[3];
    float* out = (float*)d_out;

    // Stage 1: build lane-layout bf16 A fragments (weights + bias column).
    prep_afrag<<<1, 32, 0, stream>>>(wts, bias, (unsigned short*)d_ws);

    // Stage 2: fused conv-WMMA + stencil combine.
    dim3 grid(NN / JT, NN);   // x = j-tile, y = i (D)
    sm_block3d_wmma<<<grid, 256, 0, stream>>>(img, x, (const uint4*)d_ws, out);
}
